// CausalTransformer_71906342469950
// MI455X (gfx1250) — matrix operand order repaired and checked by reading, not compiled, not run on hardware
//
#include <hip/hip_runtime.h>
#include <math.h>

// ---------------------------------------------------------------------------
// Model constants (match the JAX reference)
// ---------------------------------------------------------------------------
#define BATCH 2
#define SEQ   1024
#define NTOK  (BATCH * SEQ)        // 2048
#define HDIM  768
#define NHEAD 12
#define HEADD 64
#define FFDIM 3072
#define VOCAB 32000
#define NLAYER 4
#define TOPK  307
#define LN_EPS 1e-5f

// ---------------------------------------------------------------------------
// WMMA types (CDNA5: v_wmma_f32_16x16x32_bf16, wave32)
// ---------------------------------------------------------------------------
typedef __attribute__((ext_vector_type(16))) __bf16 v16bf;
typedef __attribute__((ext_vector_type(8)))  float  v8f;

union FragBF {
    v16bf v;
    unsigned short u[16];
    unsigned       d[8];
};

// Two f32 -> packed bf16 (RNE) in one VALU op.
__device__ __forceinline__ unsigned cvt_pk_bf16(float a, float b) {
    unsigned r;
    asm("v_cvt_pk_bf16_f32 %0, %1, %2" : "=v"(r) : "v"(a), "v"(b));
    return r;
}

__device__ __forceinline__ uint2 pack4_bf16(float a, float b, float c, float d) {
    uint2 r;
    r.x = cvt_pk_bf16(a, b);
    r.y = cvt_pk_bf16(c, d);
    return r;
}

__device__ __forceinline__ float gelu_exact(float x) {
    return 0.5f * x * (1.0f + erff(x * 0.70710678118654752f));
}

// A-matrix (16x32 bf16) lane layout inversion: k -> (half, elem)
__device__ __forceinline__ int a_half_of_k(int k) { return (k >> 3) & 1; }
__device__ __forceinline__ int a_elem_of_k(int k) { return (k & 7) | ((k >> 4) << 3); }

// ---------------------------------------------------------------------------
// Embedding: x[t, :] = tok_emb[ids[t], :] + pos_emb[t % SEQ, :]
// ---------------------------------------------------------------------------
__global__ void embed_kernel(const int* __restrict__ ids,
                             const float* __restrict__ tok,
                             const float* __restrict__ pos,
                             float* __restrict__ x) {
    const int t  = blockIdx.x;
    const int id = ids[t];
    const int s  = t & (SEQ - 1);
#pragma unroll
    for (int i = 0; i < 3; ++i) {
        const int c = threadIdx.x + i * 256;
        x[(size_t)t * HDIM + c] = tok[(size_t)id * HDIM + c] + pos[(size_t)s * HDIM + c];
    }
}

// ---------------------------------------------------------------------------
// LayerNorm over H=768. grid = NTOK, block = 256 (3 elems/thread)
// ---------------------------------------------------------------------------
__global__ void ln_kernel(const float* __restrict__ in,
                          const float* __restrict__ g,
                          const float* __restrict__ b,
                          float* __restrict__ out) {
    __shared__ float red[256];
    const int row = blockIdx.x;
    const int tid = threadIdx.x;
    const size_t base = (size_t)row * HDIM;

    const float v0 = in[base + tid];
    const float v1 = in[base + tid + 256];
    const float v2 = in[base + tid + 512];

    red[tid] = v0 + v1 + v2;
    __syncthreads();
    for (int s = 128; s > 0; s >>= 1) {
        if (tid < s) red[tid] += red[tid + s];
        __syncthreads();
    }
    const float mean = red[0] * (1.0f / HDIM);
    __syncthreads();

    const float d0 = v0 - mean, d1 = v1 - mean, d2 = v2 - mean;
    red[tid] = d0 * d0 + d1 * d1 + d2 * d2;
    __syncthreads();
    for (int s = 128; s > 0; s >>= 1) {
        if (tid < s) red[tid] += red[tid + s];
        __syncthreads();
    }
    const float rstd = rsqrtf(red[0] * (1.0f / HDIM) + LN_EPS);

    out[base + tid]       = d0 * rstd * g[tid]       + b[tid];
    out[base + tid + 256] = d1 * rstd * g[tid + 256] + b[tid + 256];
    out[base + tid + 512] = d2 * rstd * g[tid + 512] + b[tid + 512];
}

// ---------------------------------------------------------------------------
// WMMA GEMM: C[M,N] = act(A[M,K] @ B[K,N] + bias[N] (+ residual[M,N]))
// Block tile 128(M) x 64(N), K-chunk 32, double-buffered LDS (1 barrier per
// chunk). 8 waves, 2x2 register tiles -> 4 wmma per chunk per wave.
// Staging: float4 / coalesced b32 loads + v_cvt_pk_bf16_f32 + ds_store_b64
// into fragment-major swizzle; fragment reads are ds_load_b128 pairs.
// ---------------------------------------------------------------------------
__global__ void gemm_bf16_kernel(const float* __restrict__ A,
                                 const float* __restrict__ B,
                                 const float* __restrict__ bias,
                                 const float* __restrict__ residual,
                                 float* __restrict__ C,
                                 int N, int K, int act) {
    __shared__ __attribute__((aligned(32))) unsigned short lds_a[2][8 * 32 * 16];
    __shared__ __attribute__((aligned(32))) unsigned short lds_b[2][4 * 32 * 16];

    const int tid  = threadIdx.x;
    const int bm   = blockIdx.y * 128;
    const int bn   = blockIdx.x * 64;
    const int w    = tid >> 5;
    const int lane = tid & 31;
    const int half = lane >> 4;
    const int l16  = lane & 15;
    const int wm   = w >> 1;          // 0..3
    const int wn   = w & 1;           // 0..1

    // ---- loop-invariant staging addresses ----
    const float* aPtr[4];
    int aLds[4];
#pragma unroll
    for (int i = 0; i < 4; ++i) {
        const int fi = tid + (i << 8);            // 0..1023
        const int r  = fi >> 3, q4 = fi & 7;
        const int k0 = q4 << 2;
        aLds[i] = (((((r >> 4) << 5) + (a_half_of_k(k0) << 4) + (r & 15)) << 4) + a_elem_of_k(k0));
        aPtr[i] = A + (size_t)(bm + r) * K + k0;
    }
    const float* bPtr[2];
    int bLds[2];
#pragma unroll
    for (int g = 0; g < 2; ++g) {
        const int fi  = tid + (g << 8);           // 0..511
        const int col = fi & 63;
        const int jg  = fi >> 6;                  // 0..7
        const int k0  = jg << 2;
        bLds[g] = ((((col >> 4) << 5) + (a_half_of_k(k0) << 4) + (col & 15)) << 4) + a_elem_of_k(k0);
        bPtr[g] = B + (size_t)k0 * N + bn + col;
    }

    v8f acc00 = {}, acc01 = {}, acc10 = {}, acc11 = {};

    auto stage = [&](int buf, int kkc) {
#pragma unroll
        for (int i = 0; i < 4; ++i) {
            const float4 a4 = *(const float4*)(aPtr[i] + kkc);
            *(uint2*)&lds_a[buf][aLds[i]] = pack4_bf16(a4.x, a4.y, a4.z, a4.w);
        }
#pragma unroll
        for (int g = 0; g < 2; ++g) {
            const float* bp = bPtr[g] + (size_t)kkc * N;
            const float b0 = bp[0];
            const float b1 = bp[(size_t)N];
            const float b2 = bp[(size_t)2 * N];
            const float b3 = bp[(size_t)3 * N];
            *(uint2*)&lds_b[buf][bLds[g]] = pack4_bf16(b0, b1, b2, b3);
        }
    };
    auto compute = [&](int buf) {
        FragBF fa0, fa1, fb0, fb1;
        fa0.v = *(const v16bf*)&lds_a[buf][(((wm * 2 + 0) << 5) + lane) << 4];
        fa1.v = *(const v16bf*)&lds_a[buf][(((wm * 2 + 1) << 5) + lane) << 4];
        fb0.v = *(const v16bf*)&lds_b[buf][(((wn * 2 + 0) << 5) + lane) << 4];
        fb1.v = *(const v16bf*)&lds_b[buf][(((wn * 2 + 1) << 5) + lane) << 4];
        acc00 = __builtin_amdgcn_wmma_f32_16x16x32_bf16(false, fa0.v, false, fb0.v, (short)0, acc00, false, false);
        acc01 = __builtin_amdgcn_wmma_f32_16x16x32_bf16(false, fa0.v, false, fb1.v, (short)0, acc01, false, false);
        acc10 = __builtin_amdgcn_wmma_f32_16x16x32_bf16(false, fa1.v, false, fb0.v, (short)0, acc10, false, false);
        acc11 = __builtin_amdgcn_wmma_f32_16x16x32_bf16(false, fa1.v, false, fb1.v, (short)0, acc11, false, false);
    };

    for (int kk = 0; kk < K; kk += 64) {
        stage(0, kk);
        __syncthreads();
        compute(0);
        stage(1, kk + 32);
        __syncthreads();
        compute(1);
    }

    // Epilogue. C/D layout: vgpr r -> M (+8 for lanes 16-31), lane%16 -> N.
    const int ncol0 = bn + wn * 32 + l16;
    const int ncol1 = ncol0 + 16;
    const float bv0 = bias[ncol0];
    const float bv1 = bias[ncol1];
#pragma unroll
    for (int r = 0; r < 8; ++r) {
        const int mrow0 = bm + wm * 32 + half * 8 + r;
        const int mrow1 = mrow0 + 16;
        float v;
        size_t idx;

        idx = (size_t)mrow0 * N + ncol0; v = acc00[r] + bv0;
        if (residual) v += residual[idx];
        if (act) v = gelu_exact(v);
        C[idx] = v;

        idx = (size_t)mrow0 * N + ncol1; v = acc01[r] + bv1;
        if (residual) v += residual[idx];
        if (act) v = gelu_exact(v);
        C[idx] = v;

        idx = (size_t)mrow1 * N + ncol0; v = acc10[r] + bv0;
        if (residual) v += residual[idx];
        if (act) v = gelu_exact(v);
        C[idx] = v;

        idx = (size_t)mrow1 * N + ncol1; v = acc11[r] + bv1;
        if (residual) v += residual[idx];
        if (act) v = gelu_exact(v);
        C[idx] = v;
    }
}

// ---------------------------------------------------------------------------
// WMMA flash attention. One block = (batch, head, 64-row q block), 256 thr.
// S = Q @ K^T (K frags direct from global), online softmax writes P into
// bf16 A-fragment layout, O += P @ V with accumulator rescale.
// ---------------------------------------------------------------------------
__global__ void attn_kernel(const float* __restrict__ q,
                            const float* __restrict__ k,
                            const float* __restrict__ v,
                            float* __restrict__ o) {
    __shared__ __attribute__((aligned(32))) unsigned short qfrag[8 * 32 * 16];
    __shared__ __attribute__((aligned(32))) unsigned short vfrag[4 * 32 * 16];
    __shared__ __attribute__((aligned(32))) unsigned short pfrag[4 * 32 * 16];
    __shared__ float sc[64 * 33];
    __shared__ float rmax[64], rsum[64], rscale[64];

    const int tid  = threadIdx.x;
    const int qb   = blockIdx.x;
    const int h    = blockIdx.y;
    const int b    = blockIdx.z;
    const int lane = tid & 31;
    const int w    = tid >> 5;
    const int half = lane >> 4;
    const int l16  = lane & 15;

    const size_t tokbase = (size_t)b * SEQ;
    const size_t qrow0   = tokbase + (size_t)qb * 64;

    // Stage Q (x 1/8) into A-fragment layout.
#pragma unroll
    for (int i = 0; i < 4; ++i) {
        const int fi = tid + (i << 8);            // 0..1023 float4 groups
        const int rq = fi >> 4, d4 = fi & 15;
        const int d0 = d4 << 2;
        const int kc = d4 >> 3;
        const int kq0 = d0 & 31;
        const int tm = rq >> 4, m = rq & 15;
        const int addr = (((((tm * 2 + kc) << 5) + ((kq0 >> 4) << 4) + m) << 4) + (kq0 & 15));
        const float4 q4 = *(const float4*)(q + (qrow0 + rq) * HDIM + (size_t)h * HEADD + d0);
        *(uint2*)&qfrag[addr] = pack4_bf16(q4.x * 0.125f, q4.y * 0.125f, q4.z * 0.125f, q4.w * 0.125f);
    }
    if (tid < 64) { rmax[tid] = -INFINITY; rsum[tid] = 0.0f; }

    // V staging invariants: 2 groups (dim d, 4 consecutive keys j0..j0+3)
    int vLds[2];
    size_t vOff[2];
#pragma unroll
    for (int g = 0; g < 2; ++g) {
        const int fi = tid + (g << 8);            // 0..511
        const int d  = fi & 63;
        const int jg = fi >> 6;                   // 0..7
        const int j0 = jg << 2;
        vLds[g] = ((((d >> 4) << 5) + ((j0 >> 4) << 4) + (d & 15)) << 4) + (j0 & 15);
        vOff[g] = (size_t)j0 * HDIM + (size_t)h * HEADD + d;
    }

    v8f accO0 = {}, accO1 = {};
    const int otm = w >> 2;           // A tiles otm and otm+2
    const int otn = w & 3;            // V dim tile
    const int stm = w >> 1;           // score tile row
    const int stn = w & 1;            // score tile col

    __syncthreads();

    const int nkv = 2 * qb + 2;
    for (int kb = 0; kb < nkv; ++kb) {
        const size_t krow0 = tokbase + (size_t)kb * 32;
        const float* vbase = v + krow0 * HDIM;

        // Stage V tile into B-fragment layout (packed b64 stores).
#pragma unroll
        for (int g = 0; g < 2; ++g) {
            const float* vp = vbase + vOff[g];
            const float f0 = vp[0];
            const float f1 = vp[(size_t)HDIM];
            const float f2 = vp[(size_t)2 * HDIM];
            const float f3 = vp[(size_t)3 * HDIM];
            *(uint2*)&vfrag[vLds[g]] = pack4_bf16(f0, f1, f2, f3);
        }

        // Scores: one 16x16 tile per wave, K=64 in 2 chunks.
        v8f accS = {};
#pragma unroll
        for (int c = 0; c < 2; ++c) {
            FragBF fa, fbk;
            fa.v = *(const v16bf*)&qfrag[(((stm * 2 + c) << 5) + lane) << 4];
            const float4* kp4 = (const float4*)(k + (krow0 + stn * 16 + l16) * HDIM
                                                  + (size_t)h * HEADD + c * 32 + half * 16);
#pragma unroll
            for (int qd = 0; qd < 4; ++qd) {
                const float4 f4 = kp4[qd];
                fbk.d[qd * 2 + 0] = cvt_pk_bf16(f4.x, f4.y);
                fbk.d[qd * 2 + 1] = cvt_pk_bf16(f4.z, f4.w);
            }
            accS = __builtin_amdgcn_wmma_f32_16x16x32_bf16(false, fa.v, false, fbk.v, (short)0, accS, false, false);
        }
#pragma unroll
        for (int r = 0; r < 8; ++r) {
            sc[(stm * 16 + half * 8 + r) * 33 + stn * 16 + l16] = accS[r];
        }
        __syncthreads();

        // Online softmax per q row; P weights written pairwise (b32 stores).
        if (tid < 64) {
            const int qg = qb * 64 + tid;
            float mx = rmax[tid];
#pragma unroll
            for (int j = 0; j < 32; ++j) {
                if (kb * 32 + j <= qg) mx = fmaxf(mx, sc[tid * 33 + j]);
            }
            const float scale = __expf(rmax[tid] - mx);
            float sum = rsum[tid] * scale;
            const int tm = tid >> 4, mm = tid & 15;
#pragma unroll
            for (int j = 0; j < 32; j += 2) {
                const float w0 = (kb * 32 + j     <= qg) ? __expf(sc[tid * 33 + j]     - mx) : 0.0f;
                const float w1 = (kb * 32 + j + 1 <= qg) ? __expf(sc[tid * 33 + j + 1] - mx) : 0.0f;
                sum += w0 + w1;
                const int addr = (((tm << 5) + ((j >> 4) << 4) + mm) << 4) + (j & 15);
                *(unsigned*)&pfrag[addr] = cvt_pk_bf16(w0, w1);
            }
            rmax[tid] = mx; rsum[tid] = sum; rscale[tid] = scale;
        }
        __syncthreads();

        // O += P @ V (two tiles per wave), rescaled accumulators.
        FragBF pa0, pa1, fvb;
        pa0.v = *(const v16bf*)&pfrag[(((otm + 0) << 5) + lane) << 4];
        pa1.v = *(const v16bf*)&pfrag[(((otm + 2) << 5) + lane) << 4];
        fvb.v = *(const v16bf*)&vfrag[((otn << 5) + lane) << 4];
#pragma unroll
        for (int r = 0; r < 8; ++r) {
            accO0[r] *= rscale[(otm + 0) * 16 + half * 8 + r];
            accO1[r] *= rscale[(otm + 2) * 16 + half * 8 + r];
        }
        accO0 = __builtin_amdgcn_wmma_f32_16x16x32_bf16(false, pa0.v, false, fvb.v, (short)0, accO0, false, false);
        accO1 = __builtin_amdgcn_wmma_f32_16x16x32_bf16(false, pa1.v, false, fvb.v, (short)0, accO1, false, false);
        __syncthreads();
    }

    // Normalize and store O.
    const int col = otn * 16 + l16;
#pragma unroll
    for (int r = 0; r < 8; ++r) {
        const int row0 = (otm + 0) * 16 + half * 8 + r;
        const int row1 = (otm + 2) * 16 + half * 8 + r;
        o[(qrow0 + row0) * HDIM + (size_t)h * HEADD + col] = accO0[r] / rsum[row0];
        o[(qrow0 + row1) * HDIM + (size_t)h * HEADD + col] = accO1[r] / rsum[row1];
    }
}

// ---------------------------------------------------------------------------
// Top-k (k=307 of 3072) mask: exact kth-largest via 32-step binary search on
// the order-preserving uint mapping of f32; then hid *= (r >= kth).
// ---------------------------------------------------------------------------
__global__ void topk_mask_kernel(const float* __restrict__ r,
                                 float* __restrict__ hid) {
    __shared__ unsigned keys[FFDIM];
    __shared__ unsigned s_cnt;
    const int tid = threadIdx.x;
    const size_t base = (size_t)blockIdx.x * FFDIM;

#pragma unroll
    for (int j = 0; j < 12; ++j) {
        const int idx = tid + (j << 8);
        const unsigned u = __float_as_uint(r[base + idx]);
        keys[idx] = (u & 0x80000000u) ? ~u : (u | 0x80000000u);
    }
    __syncthreads();

    unsigned long long lo = 0ULL, hi = 1ULL << 32;
    for (int it = 0; it < 32; ++it) {
        const unsigned mid = (unsigned)((lo + hi) >> 1);
        if (tid == 0) s_cnt = 0;
        __syncthreads();
        int c = 0;
#pragma unroll
        for (int j = 0; j < 12; ++j) c += (keys[tid + (j << 8)] >= mid) ? 1 : 0;
        atomicAdd(&s_cnt, (unsigned)c);
        __syncthreads();
        const unsigned total = s_cnt;
        __syncthreads();
        if (total >= (unsigned)TOPK) lo = mid; else hi = mid;
    }
    const unsigned kth = (unsigned)lo;

#pragma unroll
    for (int j = 0; j < 12; ++j) {
        const int idx = tid + (j << 8);
        if (keys[idx] < kth) hid[base + idx] = 0.0f;
    }
}

// ---------------------------------------------------------------------------
// Orchestration
// ---------------------------------------------------------------------------
extern "C" void kernel_launch(void* const* d_in, const int* in_sizes, int n_in,
                              void* d_out, int out_size, void* d_ws, size_t ws_size,
                              hipStream_t stream) {
    (void)in_sizes; (void)n_in; (void)out_size; (void)ws_size;

    const int*   ids  = (const int*)d_in[0];
    // d_in[1] = attention_mask (all true -> ignored)
    const float* tok  = (const float*)d_in[2];
    const float* pos  = (const float*)d_in[3];
    const float* ln1g = (const float*)d_in[4];
    const float* ln1b = (const float*)d_in[5];
    const float* ln2g = (const float*)d_in[6];
    const float* ln2b = (const float*)d_in[7];
    const float* Wq   = (const float*)d_in[8];
    const float* bq   = (const float*)d_in[9];
    const float* Wk   = (const float*)d_in[10];
    const float* bk   = (const float*)d_in[11];
    const float* Wv   = (const float*)d_in[12];
    const float* bv   = (const float*)d_in[13];
    const float* Wo   = (const float*)d_in[14];
    const float* bo   = (const float*)d_in[15];
    const float* W1   = (const float*)d_in[16];
    const float* b1   = (const float*)d_in[17];
    const float* W2   = (const float*)d_in[18];
    const float* b2   = (const float*)d_in[19];
    const float* Wr   = (const float*)d_in[20];
    const float* br   = (const float*)d_in[21];
    const float* lnfg = (const float*)d_in[22];
    const float* lnfb = (const float*)d_in[23];
    const float* Wlm  = (const float*)d_in[24];
    const float* blm  = (const float*)d_in[25];

    // Workspace carve-up (floats)
    float* ws = (float*)d_ws;
    const size_t SH = (size_t)NTOK * HDIM;
    const size_t SF = (size_t)NTOK * FFDIM;
    float* x   = ws;
    float* xn  = x   + SH;
    float* qb_ = xn  + SH;
    float* kb_ = qb_ + SH;
    float* vb_ = kb_ + SH;
    float* ob_ = vb_ + SH;
    float* hid = ob_ + SH;
    float* rb_ = hid + SF;

    const dim3 blk256(256);
    const dim3 gRow(NTOK);
    const dim3 gGemmH(HDIM / 64, NTOK / 128);
    const dim3 gGemmF(FFDIM / 64, NTOK / 128);
    const dim3 gGemmV(VOCAB / 64, NTOK / 128);
    const dim3 gAttn(SEQ / 64, NHEAD, BATCH);

    embed_kernel<<<gRow, blk256, 0, stream>>>(ids, tok, pos, x);

    for (int l = 0; l < NLAYER; ++l) {
        const size_t oH  = (size_t)l * HDIM;
        const size_t oHH = (size_t)l * HDIM * HDIM;
        const size_t oHF = (size_t)l * HDIM * FFDIM;
        const size_t oFH = (size_t)l * FFDIM * HDIM;
        const size_t oF  = (size_t)l * FFDIM;

        // --- attention sub-block ---
        ln_kernel<<<gRow, blk256, 0, stream>>>(x, ln1g + oH, ln1b + oH, xn);
        gemm_bf16_kernel<<<gGemmH, blk256, 0, stream>>>(xn, Wq + oHH, bq + oH, nullptr, qb_, HDIM, HDIM, 0);
        gemm_bf16_kernel<<<gGemmH, blk256, 0, stream>>>(xn, Wk + oHH, bk + oH, nullptr, kb_, HDIM, HDIM, 0);
        gemm_bf16_kernel<<<gGemmH, blk256, 0, stream>>>(xn, Wv + oHH, bv + oH, nullptr, vb_, HDIM, HDIM, 0);
        attn_kernel<<<gAttn, blk256, 0, stream>>>(qb_, kb_, vb_, ob_);
        gemm_bf16_kernel<<<gGemmH, blk256, 0, stream>>>(ob_, Wo + oHH, bo + oH, x, x, HDIM, HDIM, 0);

        // --- sparse FFN sub-block ---
        ln_kernel<<<gRow, blk256, 0, stream>>>(x, ln2g + oH, ln2b + oH, xn);
        gemm_bf16_kernel<<<gGemmF, blk256, 0, stream>>>(xn, W1 + oHF, b1 + oF, nullptr, hid, FFDIM, HDIM, 1);
        gemm_bf16_kernel<<<gGemmF, blk256, 0, stream>>>(xn, Wr + oHF, br + oF, nullptr, rb_, FFDIM, HDIM, 0);
        topk_mask_kernel<<<gRow, blk256, 0, stream>>>(rb_, hid);
        gemm_bf16_kernel<<<gGemmH, blk256, 0, stream>>>(hid, W2 + oFH, b2 + oH, x, x, HDIM, FFDIM, 0);
    }

    // --- final LN + LM head ---
    ln_kernel<<<gRow, blk256, 0, stream>>>(x, lnfg, lnfb, xn);
    gemm_bf16_kernel<<<gGemmV, blk256, 0, stream>>>(xn, Wlm, blm, nullptr, (float*)d_out, VOCAB, HDIM, 0);
}
